// BeamAnchorMixtureRNN_40656160424377
// MI455X (gfx1250) — compile-verified, matrix operand order checked by reading, not run on hardware
//
#include <hip/hip_runtime.h>
#include <cstddef>

// ---------------- model dims ----------------
#define DD     512
#define MMOV   64
#define ANCH   64
#define BATCH  32
#define TENC   128
#define TDEC   64
#define TOPK   20
#define VOUT   32000
#define SCALE  0.04419417382415922f   /* 1/sqrt(512) */
#define NEG_INF (-3.402823466e38f)

typedef __attribute__((ext_vector_type(16))) _Float16 v16h;
typedef __attribute__((ext_vector_type(8)))  float    v8f;

// ============================================================
// Generic WMMA GEMM:  C[M,N] = A[M,K] @ W[N,K]^T + bias[N]
// One wave (32 threads) per 16x64 output strip (4 accumulators):
// the A fragment is loaded once per K-step and reused for 4 B
// fragments -> 4 back-to-back v_wmma per K-step.
// grid = (N/64, M/16), block = 32.  K is compile-time, mult of 32.
// Fragment layouts per CDNA5 ISA 7.12.2 (16-bit A 16x32, wave32).
// ============================================================
__device__ __forceinline__ v16h load_frag_f16(const float* __restrict__ rowptr,
                                              int kbase, int hi) {
  // lane holds K = kbase + hi*8 + {0..7}  and  kbase + 16 + hi*8 + {0..7}
  const float4* p0 = reinterpret_cast<const float4*>(rowptr + kbase + hi * 8);
  const float4* p1 = reinterpret_cast<const float4*>(rowptr + kbase + 16 + hi * 8);
  float4 a0 = p0[0], a1 = p0[1];
  float4 a2 = p1[0], a3 = p1[1];
  v16h v;
  v[0]  = (_Float16)a0.x; v[1]  = (_Float16)a0.y; v[2]  = (_Float16)a0.z; v[3]  = (_Float16)a0.w;
  v[4]  = (_Float16)a1.x; v[5]  = (_Float16)a1.y; v[6]  = (_Float16)a1.z; v[7]  = (_Float16)a1.w;
  v[8]  = (_Float16)a2.x; v[9]  = (_Float16)a2.y; v[10] = (_Float16)a2.z; v[11] = (_Float16)a2.w;
  v[12] = (_Float16)a3.x; v[13] = (_Float16)a3.y; v[14] = (_Float16)a3.z; v[15] = (_Float16)a3.w;
  return v;
}

template <int K>
__global__ void gemm_wmma(const float* __restrict__ A, int lda,
                          const float* __restrict__ W,
                          const float* __restrict__ bias,
                          float* __restrict__ C, int ldc) {
  const int tn4  = blockIdx.x;          // group of 4 consecutive 16-col tiles
  const int tm   = blockIdx.y;
  const int lane = threadIdx.x;
  const int r    = lane & 15;
  const int hi   = lane >> 4;

  const float* arow = A + (size_t)(tm * 16 + r) * (size_t)lda;
  const float* w0   = W + (size_t)(tn4 * 64 + r) * (size_t)K;
  const float* w1   = w0 + (size_t)16 * K;
  const float* w2   = w0 + (size_t)32 * K;
  const float* w3   = w0 + (size_t)48 * K;

  v8f c0 = {}, c1 = {}, c2 = {}, c3 = {};

#pragma unroll 2
  for (int kk = 0; kk < K; kk += 32) {
    if (kk + 32 < K) {
      // next K-block of the weight rows: L2 -> near cache (global_prefetch_b8)
      __builtin_prefetch(w0 + kk + 32, 0, 1);
      __builtin_prefetch(w1 + kk + 32, 0, 1);
      __builtin_prefetch(w2 + kk + 32, 0, 1);
      __builtin_prefetch(w3 + kk + 32, 0, 1);
    }
    const v16h af = load_frag_f16(arow, kk, hi);
    const v16h b0 = load_frag_f16(w0, kk, hi);
    c0 = __builtin_amdgcn_wmma_f32_16x16x32_f16(false, af, false, b0, (short)0, c0, false, false);
    const v16h b1 = load_frag_f16(w1, kk, hi);
    c1 = __builtin_amdgcn_wmma_f32_16x16x32_f16(false, af, false, b1, (short)0, c1, false, false);
    const v16h b2 = load_frag_f16(w2, kk, hi);
    c2 = __builtin_amdgcn_wmma_f32_16x16x32_f16(false, af, false, b2, (short)0, c2, false, false);
    const v16h b3 = load_frag_f16(w3, kk, hi);
    c3 = __builtin_amdgcn_wmma_f32_16x16x32_f16(false, af, false, b3, (short)0, c3, false, false);
  }

  // C tile layout (ISA 7.12.2): VGPR i -> row tm*16 + hi*8 + i, col = 16-tile base + r
#pragma unroll
  for (int nt = 0; nt < 4; ++nt) {
    const v8f  c   = (nt == 0) ? c0 : (nt == 1) ? c1 : (nt == 2) ? c2 : c3;
    const int  col = tn4 * 64 + nt * 16 + r;
    const float bb = bias ? bias[col] : 0.0f;
#pragma unroll
    for (int i = 0; i < 8; ++i) {
      const int row = tm * 16 + hi * 8 + i;
      C[(size_t)row * (size_t)ldc + col] = c[i] + bb;
    }
  }
}

// ============================================================
// Encoder: zin[b,d] = zenc[b,d] + emb_in[seq[b,t], d]
// ============================================================
__global__ void add_embed(const float* __restrict__ zenc,
                          const float* __restrict__ emb_in,
                          const int* __restrict__ seq, int t,
                          float* __restrict__ zin) {
  const int i = blockIdx.x * 256 + threadIdx.x;   // BATCH*DD
  const int d = i & (DD - 1);
  const int b = i >> 9;
  const int tok = seq[b * TENC + t];
  zin[i] = zenc[i] + emb_in[(size_t)tok * DD + d];
}

// ============================================================
// Encoder attention (65 keys) + residual + LayerNorm.
// One block per batch, 256 threads.
// proj rows: [q0 | k0 | v0] of width 3*DD.
// ============================================================
__global__ void enc_attn_ln(const float* __restrict__ zin,
                            const float* __restrict__ proj,
                            const float* __restrict__ Km,
                            const float* __restrict__ Vm,
                            const float* __restrict__ g1,
                            const float* __restrict__ b1,
                            float* __restrict__ zout) {
  const int b   = blockIdx.x;
  const int tid = threadIdx.x;
  const int wav = tid >> 5;
  const int ln  = tid & 31;

  __shared__ float s[MMOV + 1];
  __shared__ float redA[256];
  __shared__ float redB[256];
  __shared__ float mean_s, rstd_s;

  const float* q0 = proj + (size_t)b * (3 * DD);
  const float* k0 = q0 + DD;
  const float* v0 = q0 + 2 * DD;

  // 65 dot products, one wave per score
  for (int j = wav; j < MMOV + 1; j += 8) {
    const float* kv = (j == 0) ? k0 : (Km + (size_t)(j - 1) * DD);
    float p = 0.0f;
    for (int d = ln; d < DD; d += 32) p += q0[d] * kv[d];
    for (int off = 16; off; off >>= 1) p += __shfl_down(p, off);
    if (ln == 0) s[j] = p * SCALE;
  }
  __syncthreads();

  if (tid == 0) {
    float mx = s[0];
    for (int j = 1; j < MMOV + 1; ++j) mx = fmaxf(mx, s[j]);
    float sum = 0.0f;
    for (int j = 0; j < MMOV + 1; ++j) { s[j] = expf(s[j] - mx); sum += s[j]; }
    const float inv = 1.0f / sum;
    for (int j = 0; j < MMOV + 1; ++j) s[j] *= inv;
  }
  __syncthreads();

  // out = softmax @ V ; residual ; layernorm
  float v[2]; float lsum = 0.0f, lsq = 0.0f;
#pragma unroll
  for (int e = 0; e < 2; ++e) {
    const int d = tid + e * 256;
    float acc = s[0] * v0[d];
    for (int j = 0; j < MMOV; ++j) acc += s[j + 1] * Vm[(size_t)j * DD + d];
    const float val = zin[(size_t)b * DD + d] + acc;
    v[e] = val; lsum += val; lsq += val * val;
  }
  redA[tid] = lsum; redB[tid] = lsq;
  __syncthreads();
  for (int off = 128; off; off >>= 1) {
    if (tid < off) { redA[tid] += redA[tid + off]; redB[tid] += redB[tid + off]; }
    __syncthreads();
  }
  if (tid == 0) {
    const float mean = redA[0] * (1.0f / DD);
    const float var  = redB[0] * (1.0f / DD) - mean * mean;
    mean_s = mean; rstd_s = rsqrtf(var + 1e-5f);
  }
  __syncthreads();
#pragma unroll
  for (int e = 0; e < 2; ++e) {
    const int d = tid + e * 256;
    zout[(size_t)b * DD + d] = (v[e] - mean_s) * rstd_s * g1[d] + b1[d];
  }
}

// ============================================================
// Anchor gate + av LayerNorm update. grid=(ANCH, BATCH), block=256.
// ============================================================
__global__ void enc_gate_av(const float* __restrict__ Qa,
                            const float* __restrict__ ka,
                            const float* __restrict__ z,
                            float* __restrict__ av,
                            const float* __restrict__ g1,
                            const float* __restrict__ b1) {
  const int a   = blockIdx.x;
  const int b   = blockIdx.y;
  const int tid = threadIdx.x;

  __shared__ float redA[256];
  __shared__ float redB[256];
  __shared__ float gate_s, mean_s, rstd_s;

  float p = 0.0f;
  for (int d = tid; d < DD; d += 256) p += Qa[(size_t)a * DD + d] * ka[(size_t)b * DD + d];
  redA[tid] = p;
  __syncthreads();
  for (int off = 128; off; off >>= 1) {
    if (tid < off) redA[tid] += redA[tid + off];
    __syncthreads();
  }
  if (tid == 0) gate_s = 1.0f / (1.0f + expf(-redA[0] * SCALE));
  __syncthreads();
  const float g = gate_s;

  float* row = av + ((size_t)b * ANCH + a) * DD;
  float v[2]; float lsum = 0.0f, lsq = 0.0f;
#pragma unroll
  for (int e = 0; e < 2; ++e) {
    const int d = tid + e * 256;
    const float val = (1.0f - g) * row[d] + g * z[(size_t)b * DD + d];
    v[e] = val; lsum += val; lsq += val * val;
  }
  redA[tid] = lsum; redB[tid] = lsq;
  __syncthreads();
  for (int off = 128; off; off >>= 1) {
    if (tid < off) { redA[tid] += redA[tid + off]; redB[tid] += redB[tid + off]; }
    __syncthreads();
  }
  if (tid == 0) {
    const float mean = redA[0] * (1.0f / DD);
    const float var  = redB[0] * (1.0f / DD) - mean * mean;
    mean_s = mean; rstd_s = rsqrtf(var + 1e-5f);
  }
  __syncthreads();
#pragma unroll
  for (int e = 0; e < 2; ++e) {
    const int d = tid + e * 256;
    row[d] = (v[e] - mean_s) * rstd_s * g1[d] + b1[d];
  }
}

// ============================================================
// achs = concat(av[:, :ANCH-5], mover_d[:5])
// ============================================================
__global__ void build_achs(const float* __restrict__ av,
                           const float* __restrict__ mover_d,
                           float* __restrict__ achs) {
  const int i = blockIdx.x * 256 + threadIdx.x;   // BATCH*ANCH*DD
  const int d = i & (DD - 1);
  const int a = (i >> 9) & (ANCH - 1);
  achs[i] = (a < ANCH - 5) ? av[i] : mover_d[(size_t)(a - (ANCH - 5)) * DD + d];
}

// z0[b,t,:] = zenc[b,:]
__global__ void init_zdec(const float* __restrict__ zenc, float* __restrict__ z0) {
  const int i  = blockIdx.x * 256 + threadIdx.x;  // BATCH*TOPK*DD
  const int d  = i & (DD - 1);
  const int bt = i >> 9;
  const int b  = bt / TOPK;
  z0[i] = zenc[(size_t)b * DD + d];
}

// ============================================================
// Decoder step: scores -> log_softmax(TOPK axis) -> cross + aar
// -> top-20 -> aar update -> beam gather + LayerNorm.
// One block per batch, 256 threads.
// ============================================================
__global__ void dec_step(const float* __restrict__ Qdec,
                         const float* __restrict__ Kdec,
                         const float* __restrict__ disp,
                         const float* __restrict__ zcur,
                         float* __restrict__ znext,
                         float* __restrict__ aar,
                         const float* __restrict__ g2,
                         const float* __restrict__ b2) {
  const int b   = blockIdx.x;
  const int tid = threadIdx.x;

  __shared__ float sc[ANCH * TOPK];     // scores / logp, [a*TOPK + t]
  __shared__ float cr[TOPK * ANCH];     // cross, flat idx = t*ANCH + a
  __shared__ float aold[TOPK];
  __shared__ float redA[256];
  __shared__ float redB[256];
  __shared__ int   redI[256];
  __shared__ float topv[TOPK];
  __shared__ int   topi[TOPK];
  __shared__ float mean_s, rstd_s;

  if (tid < TOPK) aold[tid] = aar[b * TOPK + tid];

  // phase 1: scores sc[a,t] = SCALE * <Qdec[b,a], Kdec[b,t]>
  for (int p = tid; p < ANCH * TOPK; p += 256) {
    const int a = p / TOPK;
    const int t = p % TOPK;
    const float4* q4 = reinterpret_cast<const float4*>(Qdec + ((size_t)b * ANCH + a) * DD);
    const float4* k4 = reinterpret_cast<const float4*>(Kdec + ((size_t)b * TOPK + t) * DD);
    float acc = 0.0f;
    for (int d = 0; d < DD / 4; ++d) {
      const float4 q = q4[d], k = k4[d];
      acc += q.x * k.x + q.y * k.y + q.z * k.z + q.w * k.w;
    }
    sc[p] = acc * SCALE;
  }
  __syncthreads();

  // phase 2: log_softmax over t (TOPK) for each a
  if (tid < ANCH) {
    float mx = NEG_INF;
    for (int t = 0; t < TOPK; ++t) mx = fmaxf(mx, sc[tid * TOPK + t]);
    float sum = 0.0f;
    for (int t = 0; t < TOPK; ++t) sum += expf(sc[tid * TOPK + t] - mx);
    const float lse = mx + logf(sum);
    for (int t = 0; t < TOPK; ++t) sc[tid * TOPK + t] -= lse;
  }
  __syncthreads();

  // phase 3: cross[t*ANCH + a] = logp[a,t] + aar[t]
  for (int p = tid; p < TOPK * ANCH; p += 256) {
    const int t = p >> 6;
    const int a = p & (ANCH - 1);
    cr[p] = sc[a * TOPK + t] + aold[t];
  }
  __syncthreads();

  // phase 4: iterative top-20 (ties -> lowest flat index, like jax)
  for (int sel = 0; sel < TOPK; ++sel) {
    float bv = NEG_INF; int bi = TOPK * ANCH;
    for (int p = tid; p < TOPK * ANCH; p += 256) {
      const float v = cr[p];
      if (v > bv) { bv = v; bi = p; }
    }
    redA[tid] = bv; redI[tid] = bi;
    __syncthreads();
    for (int off = 128; off; off >>= 1) {
      if (tid < off) {
        const float v2 = redA[tid + off]; const int i2 = redI[tid + off];
        if (v2 > redA[tid] || (v2 == redA[tid] && i2 < redI[tid])) {
          redA[tid] = v2; redI[tid] = i2;
        }
      }
      __syncthreads();
    }
    if (tid == 0) { topv[sel] = redA[0]; topi[sel] = redI[0]; cr[redI[0]] = NEG_INF; }
    __syncthreads();
  }

  // phase 5: aar = val - max(val)   (topv[0] is the max)
  if (tid < TOPK) {
    float mx = topv[0];
    for (int i = 1; i < TOPK; ++i) mx = fmaxf(mx, topv[i]);
    aar[b * TOPK + tid] = topv[tid] - mx;
  }

  // phase 6: znext[b,i,:] = LN(zcur[b,beam_i,:] + disp[b,anch_i,:])
  for (int i = 0; i < TOPK; ++i) {
    const int idx  = topi[i];
    const int beam = idx >> 6;         // idx / ANCH
    const int anch = idx & (ANCH - 1); // idx % ANCH
    const float* zr = zcur + ((size_t)b * TOPK + beam) * DD;
    const float* dr = disp + ((size_t)b * ANCH + anch) * DD;
    const float v0 = zr[tid]       + dr[tid];
    const float v1 = zr[tid + 256] + dr[tid + 256];
    redA[tid] = v0 + v1;
    redB[tid] = v0 * v0 + v1 * v1;
    __syncthreads();
    for (int off = 128; off; off >>= 1) {
      if (tid < off) { redA[tid] += redA[tid + off]; redB[tid] += redB[tid + off]; }
      __syncthreads();
    }
    if (tid == 0) {
      const float mean = redA[0] * (1.0f / DD);
      const float var  = redB[0] * (1.0f / DD) - mean * mean;
      mean_s = mean; rstd_s = rsqrtf(var + 1e-5f);
    }
    __syncthreads();
    float* zo = znext + ((size_t)b * TOPK + i) * DD;
    zo[tid]       = (v0 - mean_s) * rstd_s * g2[tid]       + b2[tid];
    zo[tid + 256] = (v1 - mean_s) * rstd_s * g2[tid + 256] + b2[tid + 256];
    __syncthreads();
  }
}

// ============================================================
// Final: out[b,:] = log_softmax(logits[b,:]) over VOUT
// ============================================================
__global__ void vocab_logsoftmax(const float* __restrict__ logits,
                                 float* __restrict__ out) {
  const int b   = blockIdx.x;
  const int tid = threadIdx.x;
  __shared__ float red[256];
  __shared__ float mx_s, lse_s;
  const float* x = logits + (size_t)b * VOUT;

  float mx = NEG_INF;
  for (int j = tid; j < VOUT; j += 256) mx = fmaxf(mx, x[j]);
  red[tid] = mx;
  __syncthreads();
  for (int off = 128; off; off >>= 1) {
    if (tid < off) red[tid] = fmaxf(red[tid], red[tid + off]);
    __syncthreads();
  }
  if (tid == 0) mx_s = red[0];
  __syncthreads();

  float sum = 0.0f;
  for (int j = tid; j < VOUT; j += 256) sum += expf(x[j] - mx_s);
  red[tid] = sum;
  __syncthreads();
  for (int off = 128; off; off >>= 1) {
    if (tid < off) red[tid] += red[tid + off];
    __syncthreads();
  }
  if (tid == 0) lse_s = mx_s + logf(red[0]);
  __syncthreads();

  for (int j = tid; j < VOUT; j += 256) out[(size_t)b * VOUT + j] = x[j] - lse_s;
}

// ============================================================
// Host orchestration
// ============================================================
extern "C" void kernel_launch(void* const* d_in, const int* in_sizes, int n_in,
                              void* d_out, int out_size, void* d_ws, size_t ws_size,
                              hipStream_t stream) {
  (void)in_sizes; (void)n_in; (void)out_size; (void)ws_size;

  const int*   seq_in   = (const int*)  d_in[0];
  const float* emb_in   = (const float*)d_in[2];
  const float* vocab_W  = (const float*)d_in[4];
  const float* vocab_b  = (const float*)d_in[5];
  const float* mover_e  = (const float*)d_in[6];
  const float* anch_key = (const float*)d_in[7];
  const float* eWq = (const float*)d_in[8];  const float* eBq = (const float*)d_in[9];
  const float* eWk = (const float*)d_in[10]; const float* eBk = (const float*)d_in[11];
  const float* eWv = (const float*)d_in[12]; const float* eBv = (const float*)d_in[13];
  const float* aWq = (const float*)d_in[14]; const float* aBq = (const float*)d_in[15];
  const float* aWk = (const float*)d_in[16]; const float* aBk = (const float*)d_in[17];
  const float* dWq = (const float*)d_in[18]; const float* dBq = (const float*)d_in[19];
  const float* dWk = (const float*)d_in[20]; const float* dBk = (const float*)d_in[21];
  const float* dispW = (const float*)d_in[22]; const float* dispB = (const float*)d_in[23];
  const float* moverD = (const float*)d_in[24];
  const float* g1 = (const float*)d_in[25]; const float* b1 = (const float*)d_in[26];
  const float* g2 = (const float*)d_in[27]; const float* b2 = (const float*)d_in[28];

  float* ws = (float*)d_ws;
  size_t off = 0;
  auto alloc = [&](size_t n) { float* p = ws + off; off += n; return p; };
  float* Wqkv  = alloc((size_t)3 * DD * DD);   // packed enc Wq|Wk|Wv
  float* Bqkv  = alloc(3 * DD);
  float* Km    = alloc((size_t)MMOV * DD);
  float* Vm    = alloc((size_t)MMOV * DD);
  float* Qa    = alloc((size_t)ANCH * DD);
  float* zin   = alloc((size_t)BATCH * DD);
  float* proj  = alloc((size_t)BATCH * 3 * DD);
  float* zenc  = alloc((size_t)BATCH * DD);
  float* ka    = alloc((size_t)BATCH * DD);
  float* av    = alloc((size_t)BATCH * ANCH * DD);
  float* achs  = alloc((size_t)BATCH * ANCH * DD);
  float* disp  = alloc((size_t)BATCH * ANCH * DD);
  float* Qdec  = alloc((size_t)BATCH * ANCH * DD);
  float* zdec0 = alloc((size_t)BATCH * TOPK * DD);
  float* zdec1 = alloc((size_t)BATCH * TOPK * DD);
  float* aar   = alloc((size_t)BATCH * TOPK);
  float* Kdec  = alloc((size_t)BATCH * TOPK * DD);
  float* logit = alloc((size_t)BATCH * VOUT);

  const size_t WB = (size_t)DD * DD * sizeof(float);
  hipMemcpyAsync(Wqkv,                   eWq, WB, hipMemcpyDeviceToDevice, stream);
  hipMemcpyAsync(Wqkv + (size_t)DD*DD,   eWk, WB, hipMemcpyDeviceToDevice, stream);
  hipMemcpyAsync(Wqkv + (size_t)2*DD*DD, eWv, WB, hipMemcpyDeviceToDevice, stream);
  hipMemcpyAsync(Bqkv,        eBq, DD * sizeof(float), hipMemcpyDeviceToDevice, stream);
  hipMemcpyAsync(Bqkv + DD,   eBk, DD * sizeof(float), hipMemcpyDeviceToDevice, stream);
  hipMemcpyAsync(Bqkv + 2*DD, eBv, DD * sizeof(float), hipMemcpyDeviceToDevice, stream);
  hipMemsetAsync(zenc, 0, (size_t)BATCH * DD * sizeof(float), stream);
  hipMemsetAsync(av,   0, (size_t)BATCH * ANCH * DD * sizeof(float), stream);
  hipMemsetAsync(aar,  0, (size_t)BATCH * TOPK * sizeof(float), stream);

  // constant projections: Km/Vm (movers through enc K/V), Qa (anchor queries)
  gemm_wmma<DD><<<dim3(DD/64, MMOV/16), 32, 0, stream>>>(mover_e, DD, eWk, eBk, Km, DD);
  gemm_wmma<DD><<<dim3(DD/64, MMOV/16), 32, 0, stream>>>(mover_e, DD, eWv, eBv, Vm, DD);
  gemm_wmma<DD><<<dim3(DD/64, ANCH/16), 32, 0, stream>>>(anch_key, DD, aWq, aBq, Qa, DD);

  // -------- encoder scan --------
  for (int t = 0; t < TENC; ++t) {
    add_embed<<<(BATCH * DD) / 256, 256, 0, stream>>>(zenc, emb_in, seq_in, t, zin);
    gemm_wmma<DD><<<dim3(3*DD/64, BATCH/16), 32, 0, stream>>>(zin, DD, Wqkv, Bqkv, proj, 3*DD);
    enc_attn_ln<<<BATCH, 256, 0, stream>>>(zin, proj, Km, Vm, g1, b1, zenc);
    gemm_wmma<DD><<<dim3(DD/64, BATCH/16), 32, 0, stream>>>(zenc, DD, aWk, aBk, ka, DD);
    enc_gate_av<<<dim3(ANCH, BATCH), 256, 0, stream>>>(Qa, ka, zenc, av, g1, b1);
  }

  // -------- decoder setup --------
  build_achs<<<(BATCH * ANCH * DD) / 256, 256, 0, stream>>>(av, moverD, achs);
  gemm_wmma<DD><<<dim3(DD/64, (BATCH*ANCH)/16), 32, 0, stream>>>(achs, DD, dispW, dispB, disp, DD);
  gemm_wmma<DD><<<dim3(DD/64, (BATCH*ANCH)/16), 32, 0, stream>>>(achs, DD, dWq, dBq, Qdec, DD);
  init_zdec<<<(BATCH * TOPK * DD) / 256, 256, 0, stream>>>(zenc, zdec0);

  // -------- decoder scan --------
  float* zc = zdec0;
  float* zn = zdec1;
  for (int s = 0; s < TDEC; ++s) {
    gemm_wmma<DD><<<dim3(DD/64, (BATCH*TOPK)/16), 32, 0, stream>>>(zc, DD, dWk, dBk, Kdec, DD);
    dec_step<<<BATCH, 256, 0, stream>>>(Qdec, Kdec, disp, zc, zn, aar, g2, b2);
    float* tmp = zc; zc = zn; zn = tmp;
  }

  // -------- final vocab projection + log_softmax --------
  gemm_wmma<DD><<<dim3(VOUT/64, BATCH/16), 32, 0, stream>>>(zc, TOPK * DD, vocab_W, vocab_b,
                                                            logit, VOUT);
  vocab_logsoftmax<<<BATCH, 256, 0, stream>>>(logit, (float*)d_out);
}
